// GCN_85572928405775
// MI455X (gfx1250) — compile-verified, matrix operand order checked by reading, not run on hardware
//
#include <hip/hip_runtime.h>
#include <hip/hip_bf16.h>

#define D_IN   256
#define D_HID  128
#define N_CLS  10
#define N_GRAPHS 256
#define SLOPE  0.01f

typedef __attribute__((ext_vector_type(16))) __bf16 v16bf;
typedef __attribute__((ext_vector_type(8)))  float  v8f;

__device__ __forceinline__ unsigned short f32_to_bf16_rne(float f) {
    union { float f; unsigned u; } v; v.f = f;
    unsigned u = v.u;
    u += 0x7FFFu + ((u >> 16) & 1u);   // round-to-nearest-even
    return (unsigned short)(u >> 16);
}

// ---------------------------------------------------------------- utilities
__global__ void zero_f32(float* p, int n) {
    for (int i = blockIdx.x * blockDim.x + threadIdx.x; i < n; i += gridDim.x * blockDim.x)
        p[i] = 0.0f;
}

// Convert W[K][128] (f32) into fragment-native bf16 order:
// perm[((t*STEPS + s)*32 + lane)*16 + i]  holds  W[k(i,lane,s)][16*t + (lane&15)]
// with k(i) = 32*s + 8*(lane>>4) + (i<8 ? i : 8+i)   (ISA 7.12.2 B^T striping).
template <int K>
__global__ void cvt_w_perm(const float* __restrict__ in, unsigned short* __restrict__ out) {
    constexpr int STEPS = K / 32;
    int idx = blockIdx.x * blockDim.x + threadIdx.x;
    if (idx >= K * D_HID) return;
    int i    = idx & 15;
    int l    = (idx >> 4) & 31;
    int rest = idx >> 9;
    int s    = rest % STEPS;
    int t    = rest / STEPS;
    int ml = l & 15, hi = l >> 4;
    int kb = 32 * s + 8 * hi;
    int k  = (i < 8) ? (kb + i) : (kb + 8 + i);
    int n  = 16 * t + ml;
    out[idx] = f32_to_bf16_rne(in[k * D_HID + n]);
}

// ---------------------------------------------------------------- GCN norm
__global__ void deg_kernel(const int* __restrict__ col, const float* __restrict__ w,
                           float* __restrict__ deg, int E) {
    int e = blockIdx.x * blockDim.x + threadIdx.x;
    if (e < E) atomicAdd(&deg[col[e]], w[e]);
}

__global__ void dis_kernel(float* deg, int N) {
    int i = blockIdx.x * blockDim.x + threadIdx.x;
    if (i < N) { float d = deg[i]; deg[i] = (d > 0.0f) ? rsqrtf(d) : 0.0f; }
}

__global__ void norm_kernel(const int* __restrict__ row, const int* __restrict__ col,
                            const float* __restrict__ w, const float* __restrict__ dis,
                            float* __restrict__ norm, int E) {
    int e = blockIdx.x * blockDim.x + threadIdx.x;
    if (e < E) norm[e] = dis[row[e]] * w[e] * dis[col[e]];
}

// ---------------------------------------------------------------- bf16 WMMA GEMM
// out[M x 128] = A[M x K](f32) * Wperm(bf16 fragment-native), f32 accumulate.
// Block = 256 threads = 8 waves; block owns a 32-row strip of A staged in LDS
// as bf16. Wave w covers columns [16w,16w+16) and both 16-row M-tiles (shared
// B fragment -> 2 WMMA per loaded operand pair). Requires M % 32 == 0.
template <int K>
__global__ __launch_bounds__(256)
void gemm_bf16_wmma(const float* __restrict__ A, const unsigned short* __restrict__ Wb,
                    float* __restrict__ out, int M) {
    constexpr int STEPS = K / 32;
    __shared__ unsigned short As[32 * K];          // bf16 bits, row-major [32][K]
    const int tid = threadIdx.x;
    const int m0  = blockIdx.x * 32;

    for (int i = tid; i < 32 * K; i += 256) {
        int r = i / K, c = i % K;
        As[i] = f32_to_bf16_rne(A[(size_t)(m0 + r) * K + c]);
    }
    __syncthreads();

    const int wave = tid >> 5;          // N-tile index (0..7)
    const int lane = tid & 31;
    const int ml   = lane & 15;
    const int hi   = lane >> 4;
    const int ncol = wave * 16 + ml;    // global output column

    v8f acc0 = {}, acc1 = {};
    #pragma unroll
    for (int s = 0; s < STEPS; ++s) {
        // B fragment: contiguous 32B per lane (pre-permuted), 2x b128 loads
        const uint4* b = (const uint4*)Wb + ((size_t)(wave * STEPS + s) * 32 + lane) * 2;
        if (s + 1 < STEPS)
            __builtin_prefetch((const void*)(b + 64), 0, 3);   // global_prefetch_b8
        union { uint4 q[2]; v16bf v; } bu;
        bu.q[0] = b[0];
        bu.q[1] = b[1];

        // A fragments from LDS (16-bit A 16x32 striping, ISA 7.12.2)
        const int kb = s * 32 + hi * 8;
        union { unsigned short u[16]; v16bf v; } a0, a1;
        #pragma unroll
        for (int i = 0; i < 8; ++i) {
            a0.u[i]     = As[ml * K + kb + i];
            a0.u[i + 8] = As[ml * K + kb + 16 + i];
            a1.u[i]     = As[(16 + ml) * K + kb + i];
            a1.u[i + 8] = As[(16 + ml) * K + kb + 16 + i];
        }
        acc0 = __builtin_amdgcn_wmma_f32_16x16x32_bf16(
                   false, a0.v, false, bu.v, (short)0, acc0, false, false);
        acc1 = __builtin_amdgcn_wmma_f32_16x16x32_bf16(
                   false, a1.v, false, bu.v, (short)0, acc1, false, false);
    }

    // C/D layout: VGPR vv -> row (vv + 8*hi), col = lane&15  (no guards: M%32==0)
    #pragma unroll
    for (int vv = 0; vv < 8; ++vv) {
        int m = m0 + vv + 8 * hi;
        out[(size_t)m * D_HID + ncol]        = acc0[vv];
        out[(size_t)(m + 16) * D_HID + ncol] = acc1[vv];
    }
    (void)M;
}

// ---------------------------------------------------------------- edge scatter
// One wave per edge; lane covers 4 of 128 features.
__global__ __launch_bounds__(256)
void scatter_add(const float* __restrict__ h, const float* __restrict__ norm,
                 const int* __restrict__ row, const int* __restrict__ col,
                 float* __restrict__ agg, int E) {
    int e = (int)((blockIdx.x * blockDim.x + threadIdx.x) >> 5);
    if (e >= E) return;
    int lane = threadIdx.x & 31;
    int r = row[e], c = col[e];
    float nv = norm[e];
    const float4 hv = *(const float4*)(h + (size_t)r * D_HID + lane * 4);
    float* dst = agg + (size_t)c * D_HID + lane * 4;
    atomicAdd(dst + 0, nv * hv.x);
    atomicAdd(dst + 1, nv * hv.y);
    atomicAdd(dst + 2, nv * hv.z);
    atomicAdd(dst + 3, nv * hv.w);
}

__global__ void bias_leaky(float* __restrict__ y, const float* __restrict__ b, int total) {
    int i = blockIdx.x * blockDim.x + threadIdx.x;
    if (i < total) {
        float v = y[i] + b[i & (D_HID - 1)];
        y[i] = (v >= 0.0f) ? v : SLOPE * v;
    }
}

// ---------------------------------------------------------------- pooling
__global__ __launch_bounds__(256)
void pool_sum(const float* __restrict__ h, const int* __restrict__ batch,
              float* __restrict__ psum, float* __restrict__ pcnt, int N) {
    int n = (int)((blockIdx.x * blockDim.x + threadIdx.x) >> 5);
    if (n >= N) return;
    int lane = threadIdx.x & 31;
    int g = batch[n];
    const float4 hv = *(const float4*)(h + (size_t)n * D_HID + lane * 4);
    float* dst = psum + (size_t)g * D_HID + lane * 4;
    atomicAdd(dst + 0, hv.x);
    atomicAdd(dst + 1, hv.y);
    atomicAdd(dst + 2, hv.z);
    atomicAdd(dst + 3, hv.w);
    if (lane == 0) atomicAdd(&pcnt[g], 1.0f);
}

// ---------------------------------------------------------------- head: logits + log_softmax
__global__ __launch_bounds__(32)
void head_kernel(const float* __restrict__ psum, const float* __restrict__ pcnt,
                 const float* __restrict__ Wl, const float* __restrict__ bl,
                 float* __restrict__ out) {
    int g = blockIdx.x;
    int lane = threadIdx.x;
    __shared__ float logits[N_CLS];
    float inv = 1.0f / fmaxf(pcnt[g], 1.0f);
    if (lane < N_CLS) {
        float acc = bl[lane];
        for (int k = 0; k < D_HID; ++k)
            acc += (psum[g * D_HID + k] * inv) * Wl[k * N_CLS + lane];
        logits[lane] = acc;
    }
    __syncthreads();
    if (lane < N_CLS) {
        float m = -INFINITY;
        for (int i = 0; i < N_CLS; ++i) m = fmaxf(m, logits[i]);
        float s = 0.0f;
        for (int i = 0; i < N_CLS; ++i) s += __expf(logits[i] - m);
        out[g * N_CLS + lane] = logits[lane] - m - __logf(s);
    }
}

// ---------------------------------------------------------------- launcher
extern "C" void kernel_launch(void* const* d_in, const int* in_sizes, int n_in,
                              void* d_out, int out_size, void* d_ws, size_t ws_size,
                              hipStream_t stream) {
    const float* x    = (const float*)d_in[0];
    const int*   ei   = (const int*)  d_in[1];
    const float* ew   = (const float*)d_in[2];
    const int*   bat  = (const int*)  d_in[3];
    const float* W1   = (const float*)d_in[4];
    const float* b1   = (const float*)d_in[5];
    const float* W2   = (const float*)d_in[6];
    const float* b2   = (const float*)d_in[7];
    const float* Wl   = (const float*)d_in[8];
    const float* bl   = (const float*)d_in[9];
    float* out = (float*)d_out;

    const int N = in_sizes[0] / D_IN;      // 100000 (divisible by 32)
    const int E = in_sizes[2];             // 600000
    const int* row = ei;
    const int* col = ei + E;

    // workspace carve-up (256B aligned)
    char* base = (char*)d_ws;
    size_t off = 0;
    auto carve = [&](size_t bytes) { void* p = base + off; off = (off + bytes + 255) & ~(size_t)255; return p; };
    float*          deg  = (float*)         carve((size_t)N * 4);           // deg -> dis in place
    float*          nrm  = (float*)         carve((size_t)E * 4);
    unsigned short* W1b  = (unsigned short*)carve((size_t)D_IN * D_HID * 2);
    unsigned short* W2b  = (unsigned short*)carve((size_t)D_HID * D_HID * 2);
    float*          hbuf = (float*)         carve((size_t)N * D_HID * 4);
    float*          abuf = (float*)         carve((size_t)N * D_HID * 4);
    float*          psum = (float*)         carve((size_t)N_GRAPHS * D_HID * 4);
    float*          pcnt = (float*)         carve((size_t)N_GRAPHS * 4);
    (void)ws_size; (void)n_in; (void)out_size;

    const int T = 256;
    const int totalH = N * D_HID;

    // 1) degrees -> d^{-1/2} -> per-edge norm
    zero_f32<<<2048, T, 0, stream>>>(deg, N);
    deg_kernel<<<(E + T - 1) / T, T, 0, stream>>>(col, ew, deg, E);
    dis_kernel<<<(N + T - 1) / T, T, 0, stream>>>(deg, N);
    norm_kernel<<<(E + T - 1) / T, T, 0, stream>>>(row, col, ew, deg, nrm, E);

    // 2) weights -> fragment-native bf16
    cvt_w_perm<D_IN> <<<(D_IN * D_HID + T - 1) / T, T, 0, stream>>>(W1, W1b);
    cvt_w_perm<D_HID><<<(D_HID * D_HID + T - 1) / T, T, 0, stream>>>(W2, W2b);

    // 3) layer 1: h = x @ W1 (WMMA bf16), scatter-add with norm, bias+leaky
    gemm_bf16_wmma<D_IN><<<N / 32, T, 0, stream>>>(x, W1b, hbuf, N);
    zero_f32<<<4096, T, 0, stream>>>(abuf, totalH);
    scatter_add<<<(E + 7) / 8, T, 0, stream>>>(hbuf, nrm, row, col, abuf, E);
    bias_leaky<<<(totalH + T - 1) / T, T, 0, stream>>>(abuf, b1, totalH);

    // 4) layer 2: h = a @ W2 (WMMA bf16), scatter-add, bias+leaky
    gemm_bf16_wmma<D_HID><<<N / 32, T, 0, stream>>>(abuf, W2b, hbuf, N);
    zero_f32<<<4096, T, 0, stream>>>(abuf, totalH);
    scatter_add<<<(E + 7) / 8, T, 0, stream>>>(hbuf, nrm, row, col, abuf, E);
    bias_leaky<<<(totalH + T - 1) / T, T, 0, stream>>>(abuf, b2, totalH);

    // 5) mean pool + head
    zero_f32<<<64, T, 0, stream>>>(psum, N_GRAPHS * D_HID);
    zero_f32<<<1, T, 0, stream>>>(pcnt, N_GRAPHS);
    pool_sum<<<(N + 7) / 8, T, 0, stream>>>(abuf, bat, psum, pcnt, N);
    head_kernel<<<N_GRAPHS, 32, 0, stream>>>(psum, pcnt, Wl, bl, out);
}
